// MultiHeadAttention_80049600463162
// MI455X (gfx1250) — matrix operand order repaired and checked by reading, not compiled, not run on hardware
//
#include <hip/hip_runtime.h>

#define Bc  2
#define Sc  2048
#define Dc  1024
#define Hc  16
#define DKc 64
#define KCH 32              // K per chunk
#define NCH (Dc / KCH)      // 32 chunks
#define BROW 40             // padded LDS row stride (halves) = 80B

typedef __attribute__((ext_vector_type(16))) _Float16 v16h;
typedef __attribute__((ext_vector_type(8)))  _Float16 v8h;
typedef __attribute__((ext_vector_type(4)))  _Float16 v4h;
typedef __attribute__((ext_vector_type(8)))  float    v8f;
typedef __attribute__((ext_vector_type(4)))  float    v4f;

#if defined(__gfx1250__)
#define ASYNC_LDS 1
#else
#define ASYNC_LDS 0
#endif

// Stage 16 bytes global -> LDS. Async DMA (ASYNCcnt) on gfx1250 via inline
// asm (ISA 15.18.3 op 98: VDST = LDS byte address, VADDR = 64-bit global
// address, SADDR = off). Fallback: plain store (barrier-covered).
static __device__ __forceinline__ void stage_b128(const _Float16* g, _Float16* l) {
#if ASYNC_LDS
  unsigned lds_addr =
      (unsigned)(unsigned long long)(__attribute__((address_space(3))) _Float16*)l;
  unsigned long long gaddr = (unsigned long long)g;
  asm volatile("global_load_async_to_lds_b128 %0, %1, off"
               :: "v"(lds_addr), "v"(gaddr)
               : "memory");
#else
  *(v8h*)l = *(const v8h*)g;
#endif
}

static __device__ __forceinline__ void wait_async0() {
#if ASYNC_LDS
#if __has_builtin(__builtin_amdgcn_s_wait_asynccnt)
  __builtin_amdgcn_s_wait_asynccnt(0);
#else
  asm volatile("s_wait_asynccnt 0x0" ::: "memory");
#endif
#endif
}
static __device__ __forceinline__ void wait_async1() {
#if ASYNC_LDS
#if __has_builtin(__builtin_amdgcn_s_wait_asynccnt)
  __builtin_amdgcn_s_wait_asynccnt(1);
#else
  asm volatile("s_wait_asynccnt 0x1" ::: "memory");
#endif
#endif
}

// ---------------------------------------------------------------------------
// Elementwise fp32 -> f16 (for weights). One-time bandwidth-trivial pass.
// ---------------------------------------------------------------------------
__global__ __launch_bounds__(256)
void cvt_f32_f16_kernel(const float* __restrict__ in, _Float16* __restrict__ out,
                        int n4) {
  const int i = blockIdx.x * 256 + threadIdx.x;
  if (i < n4) {
    v4f x = *(const v4f*)(in + (size_t)i * 4);
    v4h y;
#pragma unroll
    for (int k = 0; k < 4; ++k) y[k] = (_Float16)x[k];
    *(v4h*)(out + (size_t)i * 4) = y;
  }
}

// ---------------------------------------------------------------------------
// Operand loaders (CDNA5 16-bit WMMA register layouts).
// ---------------------------------------------------------------------------
static __device__ __forceinline__ v16h load_a_f32(const float* __restrict__ row,
                                                  int k0, int half) {
  v4f a0 = *(const v4f*)(row + k0 + half * 16);
  v4f a1 = *(const v4f*)(row + k0 + half * 16 + 4);
  v4f a2 = *(const v4f*)(row + k0 + half * 16 + 8);
  v4f a3 = *(const v4f*)(row + k0 + half * 16 + 12);
  v16h r;
#pragma unroll
  for (int i = 0; i < 4; ++i) {
    r[i]      = (_Float16)a0[i];
    r[i + 4]  = (_Float16)a1[i];
    r[i + 8]  = (_Float16)a2[i];
    r[i + 12] = (_Float16)a3[i];
  }
  return r;
}

static __device__ __forceinline__ v16h load_a_f16(const _Float16* __restrict__ row,
                                                  int k0, int half) {
  v8h c0 = *(const v8h*)(row + k0 + half * 16);
  v8h c1 = *(const v8h*)(row + k0 + half * 16 + 8);
  v16h r;
#pragma unroll
  for (int i = 0; i < 8; ++i) { r[i] = c0[i]; r[i + 8] = c1[i]; }
  return r;
}

// B tile from LDS slab: lane n = column (s*16+lr); lane<16 K=0..15, >=16 K=16..31
static __device__ __forceinline__ v16h load_b_lds(const _Float16* bs, int s,
                                                  int lr, int half) {
  const _Float16* p = bs + (s * 16 + lr) * BROW + half * 16;
  v8h lo = *(const v8h*)p;
  v8h hi = *(const v8h*)(p + 8);
  v16h r;
#pragma unroll
  for (int i = 0; i < 8; ++i) { r[i] = lo[i]; r[i + 8] = hi[i]; }
  return r;
}

#define WMMA_F16(Aop, Bop, Cop) \
  __builtin_amdgcn_wmma_f32_16x16x32_f16(false, (Aop), false, (Bop), (short)0, (Cop), false, false)

// ---------------------------------------------------------------------------
// QKV projection: out = X @ W^T + b. Block = 256 M-rows x 64 N-cols; the
// weight slab for each 32-K chunk is double-buffered in LDS and filled by
// async global->LDS DMA (one b128 per thread), overlapped with WMMAs.
// transposed==0: [B,H,S,DK]; transposed==1: [B,H,DK,S].
// ---------------------------------------------------------------------------
__global__ __launch_bounds__(256)
void mha_proj_kernel(const float* __restrict__ A, const _Float16* __restrict__ Wh,
                     const float* __restrict__ bias, _Float16* __restrict__ out,
                     int transposed) {
  __shared__ _Float16 Bslab[2][64 * BROW];

  const int tid  = threadIdx.x;
  const int lane = tid & 31;
  const int warp = tid >> 5;
  const int nt   = blockIdx.x & 15;        // N-block (64 cols)
  const int mtB  = blockIdx.x >> 4;        // M-block (256 rows)
  const int half = lane >> 4;
  const int lr   = lane & 15;
  const int mwave = mtB * 8 + warp;        // this wave's 32-row tile

  // staging map: thread -> (row, 16B quarter) of the 64x32 f16 slab
  const int srow  = tid >> 2;
  const int squad = tid & 3;
  const _Float16* gW = Wh + (size_t)(nt * 64 + srow) * Dc + squad * 8;
  _Float16* lB0 = &Bslab[0][srow * BROW + squad * 8];
  _Float16* lB1 = &Bslab[1][srow * BROW + squad * 8];

  const float* arow0 = A + (size_t)(mwave * 32 + lr) * Dc;
  const float* arow1 = A + (size_t)(mwave * 32 + 16 + lr) * Dc;

  v8f acc[2][4] = {};

  stage_b128(gW, lB0);                      // chunk 0 -> buffer 0
  v16h av0 = load_a_f32(arow0, 0, half);
  v16h av1 = load_a_f32(arow1, 0, half);

  for (int i = 0; i < NCH; ++i) {
    if (i + 1 < NCH) {                      // prefetch chunk i+1 -> buffer (i+1)&1
      stage_b128(gW + (i + 1) * KCH, (i & 1) ? lB0 : lB1);
      wait_async1();                        // chunk i complete (1 still in flight)
    } else {
      wait_async0();
    }
    __syncthreads();                        // all waves' slab portions visible

    const int kn = (i + 1 < NCH) ? (i + 1) * KCH : 0;
    v16h an0 = load_a_f32(arow0, kn, half); // A for next chunk, in flight now
    v16h an1 = load_a_f32(arow1, kn, half);

    const _Float16* bs = &Bslab[i & 1][0];
    v16h bv0 = load_b_lds(bs, 0, lr, half);
    v16h bv1 = load_b_lds(bs, 1, lr, half);
    v16h bv2 = load_b_lds(bs, 2, lr, half);
    v16h bv3 = load_b_lds(bs, 3, lr, half);

    acc[0][0] = WMMA_F16(av0, bv0, acc[0][0]);
    acc[1][0] = WMMA_F16(av1, bv0, acc[1][0]);
    acc[0][1] = WMMA_F16(av0, bv1, acc[0][1]);
    acc[1][1] = WMMA_F16(av1, bv1, acc[1][1]);
    acc[0][2] = WMMA_F16(av0, bv2, acc[0][2]);
    acc[1][2] = WMMA_F16(av1, bv2, acc[1][2]);
    acc[0][3] = WMMA_F16(av0, bv3, acc[0][3]);
    acc[1][3] = WMMA_F16(av1, bv3, acc[1][3]);

    __syncthreads();                        // slab reads done before re-stage
    av0 = an0; av1 = an1;
  }

#pragma unroll
  for (int ts = 0; ts < 2; ++ts) {
#pragma unroll
    for (int s = 0; s < 4; ++s) {
      const int col = nt * 64 + s * 16 + lr;
      const float bb = bias[col];
      const int hh = col >> 6, dd = col & 63;
#pragma unroll
      for (int j = 0; j < 8; ++j) {
        const int m = mwave * 32 + ts * 16 + half * 8 + j;
        const int batch = m >> 11;
        const int srw   = m & (Sc - 1);
        size_t addr = transposed
            ? ((size_t)((batch * Hc + hh) * DKc + dd)) * Sc + srw
            : ((size_t)((batch * Hc + hh) * Sc + srw)) * DKc + dd;
        out[addr] = (_Float16)(acc[ts][s][j] + bb);
      }
    }
  }
}

// ---------------------------------------------------------------------------
// Flash attention: one wave per (b, h, 16-row q tile). Online softmax over
// 32-key chunks; causal mask computed analytically. V is pre-transposed.
// ---------------------------------------------------------------------------
__global__ __launch_bounds__(256)
void mha_attn_kernel(const _Float16* __restrict__ Q,
                     const _Float16* __restrict__ Km,
                     const _Float16* __restrict__ Vt,
                     _Float16* __restrict__ X) {
  __shared__ _Float16 Pl[8][16][32];   // per-wave P bounce (C->A relayout)

  const int lane = threadIdx.x & 31;
  const int warp = threadIdx.x >> 5;
  const int wid  = blockIdx.x * 8 + warp;
  const int qt   = wid & 127;
  const int h    = (wid >> 7) & (Hc - 1);
  const int b    = wid >> 11;
  const int half = lane >> 4;
  const int lr   = lane & 15;
  const int q0   = qt * 16;

  const _Float16* qrow = Q + ((size_t)((b * Hc + h) * Sc + q0 + lr)) * DKc;
  v16h qa[2];
#pragma unroll
  for (int kk = 0; kk < 2; ++kk) qa[kk] = load_a_f16(qrow, kk * 32, half);

  v8f o[4] = {};
  float mrun[8], lrun[8];
#pragma unroll
  for (int j = 0; j < 8; ++j) { mrun[j] = -1e30f; lrun[j] = 0.0f; }

  const _Float16* kbase = Km + ((size_t)(b * Hc + h) * Sc) * DKc;
  const _Float16* vbase = Vt + ((size_t)(b * Hc + h) * DKc) * Sc;

  for (int k0 = 0; k0 <= q0 + 15; k0 += 32) {
    float sc[2][8];
#pragma unroll
    for (int t = 0; t < 2; ++t) {
      const int key = k0 + t * 16 + lr;
      const _Float16* krow = kbase + (size_t)key * DKc;
      v8f s = {};
#pragma unroll
      for (int kk = 0; kk < 2; ++kk) {
        v16h kb = *(const v16h*)(krow + kk * 32 + half * 16);
        s = WMMA_F16(qa[kk], kb, s);
      }
#pragma unroll
      for (int j = 0; j < 8; ++j) {
        const int q = q0 + half * 8 + j;
        sc[t][j] = (key > q) ? -1e30f : s[j] * 0.125f;  // 1/sqrt(64), causal
      }
    }

#pragma unroll
    for (int j = 0; j < 8; ++j) {
      float cm = fmaxf(sc[0][j], sc[1][j]);
#pragma unroll
      for (int off = 8; off >= 1; off >>= 1)
        cm = fmaxf(cm, __shfl_xor(cm, off, 32));
      const float mt_   = fmaxf(mrun[j], cm);
      const float alpha = __expf(mrun[j] - mt_);
      const float p0 = __expf(sc[0][j] - mt_);
      const float p1 = __expf(sc[1][j] - mt_);
      float ps = p0 + p1;
#pragma unroll
      for (int off = 8; off >= 1; off >>= 1)
        ps += __shfl_xor(ps, off, 32);
      lrun[j] = lrun[j] * alpha + ps;
      mrun[j] = mt_;
#pragma unroll
      for (int s2 = 0; s2 < 4; ++s2) o[s2][j] *= alpha;
      const int row = half * 8 + j;
      Pl[warp][row][lr]      = (_Float16)p0;   // intra-wave DS, in-order
      Pl[warp][row][16 + lr] = (_Float16)p1;
    }

    v16h pa = load_a_f16(&Pl[warp][lr][0], 0, half);
#pragma unroll
    for (int s2 = 0; s2 < 4; ++s2) {
      const _Float16* vrow = vbase + (size_t)(s2 * 16 + lr) * Sc;
      v16h vb = *(const v16h*)(vrow + k0 + half * 16);
      o[s2] = WMMA_F16(pa, vb, o[s2]);
    }
  }

#pragma unroll
  for (int j = 0; j < 8; ++j) {
    const float inv = 1.0f / lrun[j];
    const int srow  = q0 + half * 8 + j;
    const size_t base = ((size_t)(b * Sc + srow)) * Dc + h * DKc;
#pragma unroll
    for (int s2 = 0; s2 < 4; ++s2)
      X[base + s2 * 16 + lr] = (_Float16)(o[s2][j] * inv);
  }
}

// ---------------------------------------------------------------------------
// Output projection: out = X @ Wo^T + bo (X f16 ws, Wh f16, out fp32 d_out).
// Same async-LDS staged structure as mha_proj_kernel.
// ---------------------------------------------------------------------------
__global__ __launch_bounds__(256)
void mha_outproj_kernel(const _Float16* __restrict__ X,
                        const _Float16* __restrict__ Wh,
                        const float* __restrict__ bias,
                        float* __restrict__ out) {
  __shared__ _Float16 Bslab[2][64 * BROW];

  const int tid  = threadIdx.x;
  const int lane = tid & 31;
  const int warp = tid >> 5;
  const int nt   = blockIdx.x & 15;
  const int mtB  = blockIdx.x >> 4;
  const int half = lane >> 4;
  const int lr   = lane & 15;
  const int mwave = mtB * 8 + warp;

  const int srow  = tid >> 2;
  const int squad = tid & 3;
  const _Float16* gW = Wh + (size_t)(nt * 64 + srow) * Dc + squad * 8;
  _Float16* lB0 = &Bslab[0][srow * BROW + squad * 8];
  _Float16* lB1 = &Bslab[1][srow * BROW + squad * 8];

  const _Float16* arow0 = X + (size_t)(mwave * 32 + lr) * Dc;
  const _Float16* arow1 = X + (size_t)(mwave * 32 + 16 + lr) * Dc;

  v8f acc[2][4] = {};

  stage_b128(gW, lB0);
  v16h av0 = load_a_f16(arow0, 0, half);
  v16h av1 = load_a_f16(arow1, 0, half);

  for (int i = 0; i < NCH; ++i) {
    if (i + 1 < NCH) {
      stage_b128(gW + (i + 1) * KCH, (i & 1) ? lB0 : lB1);
      wait_async1();
    } else {
      wait_async0();
    }
    __syncthreads();

    const int kn = (i + 1 < NCH) ? (i + 1) * KCH : 0;
    v16h an0 = load_a_f16(arow0, kn, half);
    v16h an1 = load_a_f16(arow1, kn, half);

    const _Float16* bs = &Bslab[i & 1][0];
    v16h bv0 = load_b_lds(bs, 0, lr, half);
    v16h bv1 = load_b_lds(bs, 1, lr, half);
    v16h bv2 = load_b_lds(bs, 2, lr, half);
    v16h bv3 = load_b_lds(bs, 3, lr, half);

    acc[0][0] = WMMA_F16(av0, bv0, acc[0][0]);
    acc[1][0] = WMMA_F16(av1, bv0, acc[1][0]);
    acc[0][1] = WMMA_F16(av0, bv1, acc[0][1]);
    acc[1][1] = WMMA_F16(av1, bv1, acc[1][1]);
    acc[0][2] = WMMA_F16(av0, bv2, acc[0][2]);
    acc[1][2] = WMMA_F16(av1, bv2, acc[1][2]);
    acc[0][3] = WMMA_F16(av0, bv3, acc[0][3]);
    acc[1][3] = WMMA_F16(av1, bv3, acc[1][3]);

    __syncthreads();
    av0 = an0; av1 = an1;
  }

#pragma unroll
  for (int ts = 0; ts < 2; ++ts) {
#pragma unroll
    for (int s = 0; s < 4; ++s) {
      const int col = nt * 64 + s * 16 + lr;
      const float bb = bias[col];
#pragma unroll
      for (int j = 0; j < 8; ++j) {
        const int m = mwave * 32 + ts * 16 + half * 8 + j;
        out[(size_t)m * Dc + col] = acc[ts][s][j] + bb;
      }
    }
  }
}

// ---------------------------------------------------------------------------
extern "C" void kernel_launch(void* const* d_in, const int* in_sizes, int n_in,
                              void* d_out, int out_size, void* d_ws, size_t ws_size,
                              hipStream_t stream) {
  (void)in_sizes; (void)n_in; (void)out_size; (void)ws_size;
  const float* query = (const float*)d_in[0];
  const float* keyi  = (const float*)d_in[1];
  const float* value = (const float*)d_in[2];
  // d_in[3] = causal mask: handled analytically, never read.
  const float* Wq = (const float*)d_in[4];
  const float* bq = (const float*)d_in[5];
  const float* Wk = (const float*)d_in[6];
  const float* bk = (const float*)d_in[7];
  const float* Wv = (const float*)d_in[8];
  const float* bv = (const float*)d_in[9];
  const float* Wo = (const float*)d_in[10];
  const float* bo = (const float*)d_in[11];
  float* out = (float*)d_out;

  const size_t actElems = (size_t)Bc * Hc * Sc * DKc;  // 4M f16
  const size_t wElems   = (size_t)Dc * Dc;             // 1M f16
  _Float16* ws  = (_Float16*)d_ws;
  _Float16* Qh  = ws;
  _Float16* Kh  = Qh + actElems;
  _Float16* Vt  = Kh + actElems;
  _Float16* Xh  = Vt + actElems;
  _Float16* Wqh = Xh + actElems;
  _Float16* Wkh = Wqh + wElems;
  _Float16* Wvh = Wkh + wElems;
  _Float16* Woh = Wvh + wElems;

  const dim3 blk(256);
  const int cvtBlocks  = (int)(wElems / 4 / 256);            // 1024
  const int gemmBlocks = (Bc * Sc / 256) * (Dc / 64);        // 16 * 16 = 256
  const int attnBlocks = (Bc * Hc * (Sc / 16)) / 8;          // 512

  cvt_f32_f16_kernel<<<cvtBlocks, blk, 0, stream>>>(Wq, Wqh, (int)(wElems / 4));
  cvt_f32_f16_kernel<<<cvtBlocks, blk, 0, stream>>>(Wk, Wkh, (int)(wElems / 4));
  cvt_f32_f16_kernel<<<cvtBlocks, blk, 0, stream>>>(Wv, Wvh, (int)(wElems / 4));
  cvt_f32_f16_kernel<<<cvtBlocks, blk, 0, stream>>>(Wo, Woh, (int)(wElems / 4));

  mha_proj_kernel<<<gemmBlocks, blk, 0, stream>>>(query, Wqh, bq, Qh, 0);
  mha_proj_kernel<<<gemmBlocks, blk, 0, stream>>>(keyi,  Wkh, bk, Kh, 0);
  mha_proj_kernel<<<gemmBlocks, blk, 0, stream>>>(value, Wvh, bv, Vt, 1);
  mha_attn_kernel<<<attnBlocks, blk, 0, stream>>>(Qh, Kh, Vt, Xh);
  mha_outproj_kernel<<<gemmBlocks, blk, 0, stream>>>(Xh, Woh, bo, out);
}
